// GroupedQueryAttention_21938692948616
// MI455X (gfx1250) — compile-verified, matrix operand order checked by reading, not run on hardware
//
#include <hip/hip_runtime.h>
#include <stdint.h>

typedef __attribute__((ext_vector_type(16))) __bf16 v16bf;
typedef __attribute__((ext_vector_type(8)))  float  v8f;

#define EMB   2048
#define S_LEN 2048
#define BATCH 2
#define NQH   16
#define NKV   4
#define HD    128
#define KVDIM (NKV*HD)   /* 512 */

union Frag16 {
  v16bf    v;
  uint16_t s[16];
  uint4    q[2];
};

static __device__ __forceinline__ uint16_t f2bf_raw(float f) {
  union { float f; uint32_t u; } a; a.f = f;
  uint32_t r = a.u + 0x7FFFu + ((a.u >> 16) & 1u);   // round-to-nearest-even
  return (uint16_t)(r >> 16);
}

// 16x32 bf16 WMMA operand fragment (ISA 16-bit A layout; identical pattern for
// the B operand when the stored matrix is row-major with rows = B's N columns).
// lanes 0-15: row = lane, K {0..7, 16..23}; lanes 16-31: row = lane-16, K {8..15, 24..31}.
static __device__ __forceinline__ v16bf load_frag(const uint16_t* base, int stride,
                                                  int row0, int k0) {
  int l  = threadIdx.x & 31;
  int r  = l & 15;
  int ko = (l & 16) ? 8 : 0;
  const uint16_t* p = base + (size_t)(row0 + r) * stride + k0 + ko;
  Frag16 f;
  f.q[0] = *(const uint4*)(p);
  f.q[1] = *(const uint4*)(p + 16);
  return f.v;
}

__global__ void cvt_f32_bf16(const float* __restrict__ src, uint16_t* __restrict__ dst, int n) {
  for (int i = blockIdx.x * blockDim.x + threadIdx.x; i < n; i += gridDim.x * blockDim.x)
    dst[i] = f2bf_raw(src[i]);
}

// (b, s, hkv, d) -> (b, hkv, d, s)  so PV A-fragments (Vt rows = d) are contiguous.
__global__ void transpose_v(const uint16_t* __restrict__ src, uint16_t* __restrict__ dst) {
  const int total = BATCH * S_LEN * KVDIM;
  for (int i = blockIdx.x * blockDim.x + threadIdx.x; i < total; i += gridDim.x * blockDim.x) {
    int d   = i % HD;
    int hkv = (i / HD) % NKV;
    int s   = (i / KVDIM) % S_LEN;
    int b   = i / (KVDIM * S_LEN);
    dst[(((size_t)b * NKV + hkv) * HD + d) * S_LEN + s] = src[i];
  }
}

// C[M,N] = A[M,K] * B[N,K]^T, bf16 in, f32 accumulate.
// Each wave: 32x64 tile -> 8 WMMAs per 12 b128 loads per k-step (1.5 loads/WMMA).
// One-lane-per-row prefetch 256 elements (8 k-steps) ahead covers each 64B line once.
__global__ void gemm_bf16_nt(const uint16_t* __restrict__ A,
                             const uint16_t* __restrict__ Bm,
                             float* __restrict__ Cf,
                             uint16_t* __restrict__ Cb,
                             int M, int N, int K, float scale) {
  const int wid = threadIdx.x >> 5;
  const int l   = threadIdx.x & 31;
  const int m0  = blockIdx.y * 256 + wid * 32;
  const int n0  = blockIdx.x * 64;
  v8f acc[2][4] = {};
  for (int k0 = 0; k0 < K; k0 += 32) {
    const int kpf = k0 + 256;
    if (kpf < K) {
      __builtin_prefetch(A  + (size_t)(m0 + l) * K + kpf, 0, 3);        // 32 A rows
      __builtin_prefetch(Bm + (size_t)(n0 + l) * K + kpf, 0, 3);        // B rows n0..n0+31
      __builtin_prefetch(Bm + (size_t)(n0 + 32 + l) * K + kpf, 0, 3);   // B rows n0+32..n0+63
    }
    v16bf a0 = load_frag(A, K, m0,      k0);
    v16bf a1 = load_frag(A, K, m0 + 16, k0);
    #pragma unroll
    for (int j = 0; j < 4; ++j) {
      v16bf bf = load_frag(Bm, K, n0 + j * 16, k0);
      acc[0][j] = __builtin_amdgcn_wmma_f32_16x16x32_bf16(false, a0, false, bf,
                                                          (short)0, acc[0][j], false, false);
      acc[1][j] = __builtin_amdgcn_wmma_f32_16x16x32_bf16(false, a1, false, bf,
                                                          (short)0, acc[1][j], false, false);
    }
  }
  const int rb  = (l & 16) ? 8 : 0;
  const int col = l & 15;
  #pragma unroll
  for (int half = 0; half < 2; ++half) {
    #pragma unroll
    for (int j = 0; j < 4; ++j) {
      #pragma unroll
      for (int i = 0; i < 8; ++i) {
        size_t idx = (size_t)(m0 + half * 16 + rb + i) * N + n0 + j * 16 + col;
        float v = acc[half][j][i] * scale;
        if (Cb) Cb[idx] = f2bf_raw(v);
        else    Cf[idx] = v;
      }
    }
  }
}

// Flash attention, one wave per 16-query tile of one (b, q-head).
// Computes S^T = K*Q^T so each lane owns one query column: softmax reduction is
// in-register + one shfl_xor(16), and exp(S^T) registers ARE the B operand of
// O^T = V^T * P^T with no cross-lane movement.
__global__ void flash_attn_gqa(const uint16_t* __restrict__ Q,
                               const uint16_t* __restrict__ Kc,
                               const uint16_t* __restrict__ Vt,
                               uint16_t* __restrict__ O,
                               const int* __restrict__ is_causal_p) {
  const int l    = threadIdx.x & 31;
  const int wid  = threadIdx.x >> 5;
  const int tile = blockIdx.x * (blockDim.x >> 5) + wid;
  const int qt   = tile & 127;          // S/16 query tiles per head
  const int head = tile >> 7;           // b*NQH + h
  const int h    = head & (NQH - 1);
  const int b    = head >> 4;
  const int qbase = qt * 16;
  const int hkv   = h >> 2;             // group size 4

  const uint16_t* Qb = Q  + (size_t)b * S_LEN * EMB   + h   * HD;
  const uint16_t* Kb = Kc + (size_t)b * S_LEN * KVDIM + hkv * HD;
  const uint16_t* Vb = Vt + ((size_t)(b * NKV + hkv)) * HD * S_LEN;

  const int causal = is_causal_p[0];

  v16bf qf[4];
  #pragma unroll
  for (int kk = 0; kk < 4; ++kk) qf[kk] = load_frag(Qb, EMB, qbase, kk * 32);

  v8f   oacc[8] = {};
  float m_run = -1e30f, l_run = 0.0f;

  const int   qcol = l & 15;
  const int   qg   = qbase + qcol;
  const int   hi   = (l & 16) ? 8 : 0;
  const float L2E  = 1.4426950408889634f;

  const int nkt = causal ? ((qbase + 16 + 31) >> 5) : (S_LEN >> 5);
  for (int t = 0; t < nkt; ++t) {
    const int n0 = t * 32;
    // Prefetch the key/value tile 4 iterations (128 keys) ahead.
    if (t + 4 < nkt) {
      const int npf = n0 + 128;
      #pragma unroll
      for (int c = 0; c < 4; ++c)                         // 32 K rows x 256B each
        __builtin_prefetch(Kb + (size_t)(npf + l) * KVDIM + c * 32, 0, 3);
      #pragma unroll
      for (int c = 0; c < 4; ++c)                         // 128 Vt rows x 64B each
        __builtin_prefetch(Vb + (size_t)(c * 32 + l) * S_LEN + npf, 0, 3);
    }
    v8f sa = {}, sb = {};
    #pragma unroll
    for (int kk = 0; kk < 4; ++kk) {
      v16bf ka = load_frag(Kb, KVDIM, n0,      kk * 32);
      sa = __builtin_amdgcn_wmma_f32_16x16x32_bf16(false, ka, false, qf[kk], (short)0, sa, false, false);
      v16bf kb = load_frag(Kb, KVDIM, n0 + 16, kk * 32);
      sb = __builtin_amdgcn_wmma_f32_16x16x32_bf16(false, kb, false, qf[kk], (short)0, sb, false, false);
    }
    if (causal) {
      #pragma unroll
      for (int i = 0; i < 8; ++i) {
        if (n0 + hi + i      > qg) sa[i] = -1e30f;
        if (n0 + 16 + hi + i > qg) sb[i] = -1e30f;
      }
    }
    float tmax = sa[0];
    #pragma unroll
    for (int i = 1; i < 8; ++i) tmax = fmaxf(tmax, sa[i]);
    #pragma unroll
    for (int i = 0; i < 8; ++i) tmax = fmaxf(tmax, sb[i]);
    tmax = fmaxf(tmax, __shfl_xor(tmax, 16, 32));
    const float mn    = fmaxf(m_run, tmax);
    const float alpha = exp2f((m_run - mn) * L2E);
    float pa[8], pb[8], rsum = 0.0f;
    #pragma unroll
    for (int i = 0; i < 8; ++i) { pa[i] = exp2f((sa[i] - mn) * L2E); rsum += pa[i]; }
    #pragma unroll
    for (int i = 0; i < 8; ++i) { pb[i] = exp2f((sb[i] - mn) * L2E); rsum += pb[i]; }
    rsum += __shfl_xor(rsum, 16, 32);
    l_run = l_run * alpha + rsum;
    m_run = mn;
    #pragma unroll
    for (int dd = 0; dd < 8; ++dd)
      #pragma unroll
      for (int i = 0; i < 8; ++i) oacc[dd][i] *= alpha;
    Frag16 pf;
    #pragma unroll
    for (int i = 0; i < 8; ++i) { pf.s[i] = f2bf_raw(pa[i]); pf.s[8 + i] = f2bf_raw(pb[i]); }
    #pragma unroll
    for (int dd = 0; dd < 8; ++dd) {
      v16bf vf = load_frag(Vb, S_LEN, dd * 16, n0);
      oacc[dd] = __builtin_amdgcn_wmma_f32_16x16x32_bf16(false, vf, false, pf.v, (short)0, oacc[dd], false, false);
    }
  }
  const float inv = 1.0f / l_run;
  // O^T frag: reg i -> d = dd*16 + i (+8 for hi lanes), column = query (lane%16)
  uint16_t* Ob = O + (size_t)(b * S_LEN + qbase + qcol) * EMB + h * HD;
  #pragma unroll
  for (int dd = 0; dd < 8; ++dd)
    #pragma unroll
    for (int i = 0; i < 8; ++i)
      Ob[dd * 16 + hi + i] = f2bf_raw(oacc[dd][i] * inv);
}

extern "C" void kernel_launch(void* const* d_in, const int* in_sizes, int n_in,
                              void* d_out, int out_size, void* d_ws, size_t ws_size,
                              hipStream_t stream) {
  const float* x  = (const float*)d_in[0];
  // d_in[1] = attn_mask (unused placeholder)
  const int*   is_causal = (const int*)d_in[2];
  const float* Wq = (const float*)d_in[3];
  const float* Wk = (const float*)d_in[4];
  const float* Wv = (const float*)d_in[5];
  const float* Wo = (const float*)d_in[6];
  float* out = (float*)d_out;

  const size_t nX  = (size_t)BATCH * S_LEN * EMB;    // 8388608
  const size_t nWq = (size_t)EMB * EMB;              // 4194304
  const size_t nWk = (size_t)KVDIM * EMB;            // 1048576
  const size_t nKV = (size_t)BATCH * S_LEN * KVDIM;  // 2097152

  uint16_t* ws  = (uint16_t*)d_ws;
  uint16_t* xb  = ws;  ws += nX;     // bf16 x      — reused as attention output later
  uint16_t* wqb = ws;  ws += nWq;
  uint16_t* wkb = ws;  ws += nWk;
  uint16_t* wvb = ws;  ws += nWk;
  uint16_t* wob = ws;  ws += nWq;
  uint16_t* qb  = ws;  ws += nX;
  uint16_t* kb  = ws;  ws += nKV;
  uint16_t* vb  = ws;  ws += nKV;
  uint16_t* vtb = ws;  ws += nKV;
  uint16_t* ab  = xb;                // alias: xb is dead after V projection

  cvt_f32_bf16<<<2048, 256, 0, stream>>>(x,  xb,  (int)nX);
  cvt_f32_bf16<<<2048, 256, 0, stream>>>(Wq, wqb, (int)nWq);
  cvt_f32_bf16<<<2048, 256, 0, stream>>>(Wk, wkb, (int)nWk);
  cvt_f32_bf16<<<2048, 256, 0, stream>>>(Wv, wvb, (int)nWk);
  cvt_f32_bf16<<<2048, 256, 0, stream>>>(Wo, wob, (int)nWq);

  const int   M = BATCH * S_LEN;                 // 4096
  const float qscale = 0.08838834764831845f;     // 1/sqrt(HD), folded into Q

  gemm_bf16_nt<<<dim3(EMB/64,   M/256), 256, 0, stream>>>(xb, wqb, nullptr, qb, M, EMB,   EMB, qscale);
  gemm_bf16_nt<<<dim3(KVDIM/64, M/256), 256, 0, stream>>>(xb, wkb, nullptr, kb, M, KVDIM, EMB, 1.0f);
  gemm_bf16_nt<<<dim3(KVDIM/64, M/256), 256, 0, stream>>>(xb, wvb, nullptr, vb, M, KVDIM, EMB, 1.0f);

  transpose_v<<<2048, 256, 0, stream>>>(vb, vtb);

  // B*NQH*(S/16) = 4096 wave-tiles, 4 waves per 128-thread block
  flash_attn_gqa<<<1024, 128, 0, stream>>>(qb, kb, vtb, ab, is_causal);

  gemm_bf16_nt<<<dim3(EMB/64, M/256), 256, 0, stream>>>(ab, wob, out, nullptr, M, EMB, EMB, 1.0f);
}
